// SelfAttention2D_78211354460527
// MI455X (gfx1250) — matrix operand order repaired and checked by reading, not compiled, not run on hardware
//
#include <hip/hip_runtime.h>

// ---------------------------------------------------------------------------
// SelfAttention2D on MI455X (gfx1250): fused flash-attention, f16 WMMA path,
// double-buffered async global->LDS staging (ASYNCcnt), coalesced epilogue.
// B=8, C=64, H=W=64, L=4096. scale = 1/sqrt(L) = 1/64, folded into q.
// ---------------------------------------------------------------------------

typedef __attribute__((ext_vector_type(8)))  float    v8f;
typedef __attribute__((ext_vector_type(16))) _Float16 v16h;

union Op16 {           // one 16-bit WMMA operand (8 VGPRs)
  v16h v;
  int4 i[2];
  _Float16 h[16];
};
union Acc8 {           // one 16x16 f32 accumulator (8 VGPRs)
  v8f v;
  float f[8];
};

#define WMMA_F16(A, B, Cacc) \
  __builtin_amdgcn_wmma_f32_16x16x32_f16(false, (A), false, (B), (short)0, (Cacc), false, false)

// CDNA5 async memory->LDS copy (per-lane 16B), tracked by ASYNCcnt.
static __device__ __forceinline__ void async_copy_b128(unsigned int lds_addr,
                                                       const void* gaddr) {
  asm volatile("global_load_async_to_lds_b128 %0, %1, off"
               :: "v"(lds_addr), "v"(gaddr) : "memory");
}
#define S_WAIT_ASYNCCNT_4() asm volatile("s_wait_asynccnt 4" ::: "memory")
#define S_WAIT_ASYNCCNT_0() asm volatile("s_wait_asynccnt 0" ::: "memory")

static __device__ __forceinline__ float half_group_max(float x) {
  // reduce across a 16-lane half of the wave32 (accumulator row group)
  #pragma unroll
  for (int m = 1; m < 16; m <<= 1) x = fmaxf(x, __shfl_xor(x, m, 32));
  return x;
}
static __device__ __forceinline__ float half_group_sum(float x) {
  #pragma unroll
  for (int m = 1; m < 16; m <<= 1) x += __shfl_xor(x, m, 32);
  return x;
}

// ---------------------------------------------------------------------------
// Kernel 1: q/k/v = conv1x1(x) projections, fp32 -> f16, WMMA 16x16x32.
//   qT[B,L,C] = (Wq x + bq)^T * (1/64)   (A-operand friendly: rows = queries)
//   kT[B,L,C] = (Wk x + bk)^T            (B-operand friendly for S: K dim = c)
//   v [B,C,L] =  Wv x + bv               (B-operand friendly for O: K dim = j)
// ---------------------------------------------------------------------------

static __device__ __forceinline__ v8f proj_tile(const float* __restrict__ W,
                                                const Op16* bx, int c0, int m, int h) {
  Op16 A[2];
  #pragma unroll
  for (int a = 0; a < 2; ++a) {
    // A layout (16-bit 16x32): lane m (both halves) reads row m of W chunk:
    // VGPR0-3 hold K = h*8 + 0..7, VGPR4-7 hold K = 16 + h*8 + 0..7
    const float* wr = W + (size_t)(c0 + m) * 64 + a * 32 + h * 8;
    const float4 w0 = *(const float4*)(wr + 0);
    const float4 w1 = *(const float4*)(wr + 4);
    const float4 w2 = *(const float4*)(wr + 16);
    const float4 w3 = *(const float4*)(wr + 20);
    A[a].h[0]  = (_Float16)w0.x; A[a].h[1]  = (_Float16)w0.y;
    A[a].h[2]  = (_Float16)w0.z; A[a].h[3]  = (_Float16)w0.w;
    A[a].h[4]  = (_Float16)w1.x; A[a].h[5]  = (_Float16)w1.y;
    A[a].h[6]  = (_Float16)w1.z; A[a].h[7]  = (_Float16)w1.w;
    A[a].h[8]  = (_Float16)w2.x; A[a].h[9]  = (_Float16)w2.y;
    A[a].h[10] = (_Float16)w2.z; A[a].h[11] = (_Float16)w2.w;
    A[a].h[12] = (_Float16)w3.x; A[a].h[13] = (_Float16)w3.y;
    A[a].h[14] = (_Float16)w3.z; A[a].h[15] = (_Float16)w3.w;
  }
  v8f acc = {0.f, 0.f, 0.f, 0.f, 0.f, 0.f, 0.f, 0.f};
  acc = WMMA_F16(A[0].v, bx[0].v, acc);
  acc = WMMA_F16(A[1].v, bx[1].v, acc);
  return acc;
}

__global__ void __launch_bounds__(128)
qkv_proj_kernel(const float* __restrict__ x,
                const float* __restrict__ Wq, const float* __restrict__ bq,
                const float* __restrict__ Wk, const float* __restrict__ bk,
                const float* __restrict__ Wv, const float* __restrict__ bv,
                _Float16* __restrict__ qT, _Float16* __restrict__ kT,
                _Float16* __restrict__ vM) {
  constexpr int C = 64, L = 4096;
  const int b    = blockIdx.x >> 8;          // 256 l-tiles of 16 per batch
  const int l0   = (blockIdx.x & 255) << 4;
  const int tid  = threadIdx.x;
  const int wv   = tid >> 5;
  const int lane = tid & 31;
  const int n = lane & 15, h = lane >> 4;

  __shared__ __align__(16) _Float16 xT_s[16 * 64];   // [l][c], 128B rows

  // stage x[b, 0:64, l0:l0+16] transposed to f16
  #pragma unroll
  for (int it = 0; it < 2; ++it) {
    const int idx = tid + it * 128;                  // 0..255
    const int row = idx >> 2, seg = idx & 3;         // row = c
    const float4 d =
        *(const float4*)(x + (((size_t)(b * C + row)) << 12) + l0 + (seg << 2));
    xT_s[(seg * 4 + 0) * 64 + row] = (_Float16)d.x;
    xT_s[(seg * 4 + 1) * 64 + row] = (_Float16)d.y;
    xT_s[(seg * 4 + 2) * 64 + row] = (_Float16)d.z;
    xT_s[(seg * 4 + 3) * 64 + row] = (_Float16)d.w;
  }
  __syncthreads();

  // B operands (x tile), shared by q/k/v: lane (n,h) reads K = h*8+0..7, 16+h*8+0..7
  Op16 bx[2];
  #pragma unroll
  for (int a = 0; a < 2; ++a) {
    const int off = n * 64 + a * 32 + h * 8;
    bx[a].i[0] = *(const int4*)&xT_s[off];
    bx[a].i[1] = *(const int4*)&xT_s[off + 16];
  }

  const int c0 = wv << 4;   // 16 output channels per wave
  const int m  = n;

  // ---- q (pre-scaled by 1/sqrt(L)) ----
  {
    Acc8 acc; acc.v = proj_tile(Wq, bx, c0, m, h);
    #pragma unroll
    for (int r = 0; r < 8; ++r) {
      const int c = c0 + r + 8 * h;
      const float val = (acc.f[r] + bq[c]) * 0.015625f;
      qT[(((size_t)(b * L + l0 + n)) << 6) + c] = (_Float16)val;
    }
  }
  // ---- k ----
  {
    Acc8 acc; acc.v = proj_tile(Wk, bx, c0, m, h);
    #pragma unroll
    for (int r = 0; r < 8; ++r) {
      const int c = c0 + r + 8 * h;
      kT[(((size_t)(b * L + l0 + n)) << 6) + c] = (_Float16)(acc.f[r] + bk[c]);
    }
  }
  // ---- v ----
  {
    Acc8 acc; acc.v = proj_tile(Wv, bx, c0, m, h);
    #pragma unroll
    for (int r = 0; r < 8; ++r) {
      const int c = c0 + r + 8 * h;
      vM[(((size_t)(b * C + c)) << 12) + l0 + n] = (_Float16)(acc.f[r] + bv[c]);
    }
  }
}

// ---------------------------------------------------------------------------
// Kernel 2: fused flash attention + gamma residual.
// Block = 128 threads (4 waves); each wave owns 16 query rows, Q in registers.
// Per 32-key tile: async double-buffered k/v staging, 4 WMMAs for S (16x32),
// online softmax (half-wave shfl reductions match the 16x16 C-layout:
// VGPR r <-> rows r / r+8), P bounced through LDS (C-layout -> A-layout),
// 4 WMMAs for O += P*V^T. Epilogue transposes O via LDS for coalesced stores.
// ---------------------------------------------------------------------------

// per-thread share of one tile stage: 4 async b128 instructions
static __device__ __forceinline__ void stage_tiles(const _Float16* __restrict__ kT,
                                                   const _Float16* __restrict__ vM,
                                                   _Float16* k_s, _Float16* v_s,
                                                   int b, int j0, int buf, int tid) {
  #pragma unroll
  for (int it = 0; it < 2; ++it) {
    const int idx = tid + it * 128;            // 0..255 int4 slots
    const int row = idx >> 3, ch = idx & 7;    // k_s: 32 rows x 8 int4
    const _Float16* src = kT + (((size_t)(b * 4096 + j0 + row)) << 6) + (ch << 3);
    async_copy_b128((unsigned int)(size_t)(k_s + buf * 2048 + idx * 8), src);
  }
  #pragma unroll
  for (int it = 0; it < 2; ++it) {
    const int idx = tid + it * 128;
    const int row = idx >> 2, ch = idx & 3;    // v_s: 64 rows x 4 int4
    const _Float16* src = vM + (((size_t)(b * 64 + row)) << 12) + j0 + (ch << 3);
    async_copy_b128((unsigned int)(size_t)(v_s + buf * 2048 + idx * 8), src);
  }
}

__global__ void __launch_bounds__(128)
attn_kernel(const _Float16* __restrict__ qT, const _Float16* __restrict__ kT,
            const _Float16* __restrict__ vM, const float* __restrict__ x,
            const float* __restrict__ gamma, float* __restrict__ out) {
  constexpr int C = 64, L = 4096;
  const int b    = blockIdx.x >> 6;          // 64 query tiles of 64 per batch
  const int q0   = (blockIdx.x & 63) << 6;
  const int tid  = threadIdx.x;
  const int wv   = tid >> 5;
  const int lane = tid & 31;
  const int n = lane & 15, h = lane >> 4;
  const int i0 = q0 + (wv << 4);             // this wave's query base

  // 20 KB shared, manually carved; o_s reuses the staging region after loop.
  __shared__ __align__(16) unsigned char smem[20480];
  _Float16* k_s = (_Float16*)smem;                    // 2 x [32][64] f16, 8 KB
  _Float16* v_s = (_Float16*)(smem + 8192);           // 2 x [64][32] f16, 8 KB
  _Float16* p_s = (_Float16*)(smem + 16384);          // 4 x [16][32] f16, 4 KB
  float*    o_s = (float*)smem;                       // [64][65] f32 (epilogue)

  // Q A-operands (two K-chunks of 32 channels), resident for the whole loop
  Op16 qa[2];
  {
    const _Float16* qrow = qT + (((size_t)(b * L + i0 + n)) << 6);
    #pragma unroll
    for (int a = 0; a < 2; ++a) {
      qa[a].i[0] = *(const int4*)(qrow + a * 32 + h * 8);
      qa[a].i[1] = *(const int4*)(qrow + a * 32 + 16 + h * 8);
    }
  }

  Acc8 O[4];
  #pragma unroll
  for (int t = 0; t < 4; ++t) {
    v8f z = {0.f, 0.f, 0.f, 0.f, 0.f, 0.f, 0.f, 0.f};
    O[t].v = z;
  }
  float mrow[8], lrow[8];
  #pragma unroll
  for (int r = 0; r < 8; ++r) { mrow[r] = -INFINITY; lrow[r] = 0.f; }

  // prefetch tile 0 into buffer 0
  stage_tiles(kT, vM, k_s, v_s, b, 0, 0, tid);

  for (int jt = 0; jt < 128; ++jt) {
    const int buf = jt & 1;
    if (jt + 1 < 128) {
      // prefetch next tile into the other buffer (its readers finished at the
      // end-of-previous-iteration barrier), then wait for the current tile:
      // async loads complete in order, so ASYNCcnt <= 4 => current tile done.
      stage_tiles(kT, vM, k_s, v_s, b, (jt + 1) * 32, buf ^ 1, tid);
      S_WAIT_ASYNCCNT_4();
    } else {
      S_WAIT_ASYNCCNT_0();
    }
    __syncthreads();

    const _Float16* kb_s = k_s + buf * 2048;
    const _Float16* vb_s = v_s + buf * 2048;

    // ---- S = Q x K-tile : two 16x16 score tiles, K=64 in 2 steps ----
    Acc8 S[2];
    #pragma unroll
    for (int nt = 0; nt < 2; ++nt) {
      v8f acc = {0.f, 0.f, 0.f, 0.f, 0.f, 0.f, 0.f, 0.f};
      #pragma unroll
      for (int a = 0; a < 2; ++a) {
        Op16 kb;
        const int off = (nt * 16 + n) * 64 + a * 32 + h * 8;
        kb.i[0] = *(const int4*)&kb_s[off];
        kb.i[1] = *(const int4*)&kb_s[off + 16];
        acc = WMMA_F16(qa[a].v, kb.v, acc);
      }
      S[nt].v = acc;
    }

    // ---- online softmax over the 16x32 strip ----
    #pragma unroll
    for (int r = 0; r < 8; ++r) {
      const float mx   = half_group_max(fmaxf(S[0].f[r], S[1].f[r]));
      const float mnew = fmaxf(mrow[r], mx);
      const float fac  = __expf(mrow[r] - mnew);
      mrow[r] = mnew;
      const float p0 = __expf(S[0].f[r] - mnew);
      const float p1 = __expf(S[1].f[r] - mnew);
      lrow[r] = lrow[r] * fac + half_group_sum(p0 + p1);
      #pragma unroll
      for (int t = 0; t < 4; ++t) O[t].f[r] *= fac;
      // spill P (C-layout) to LDS; element (row r+8h, col n / 16+n)
      p_s[wv * 512 + (r + 8 * h) * 32 + n]      = (_Float16)p0;
      p_s[wv * 512 + (r + 8 * h) * 32 + 16 + n] = (_Float16)p1;
    }

    // ---- reload P in A-layout, O += P x V^T (4 channel tiles) ----
    Op16 pa;
    pa.i[0] = *(const int4*)&p_s[wv * 512 + n * 32 + h * 8];
    pa.i[1] = *(const int4*)&p_s[wv * 512 + n * 32 + 16 + h * 8];
    #pragma unroll
    for (int t = 0; t < 4; ++t) {
      Op16 vb;
      const int off = (t * 16 + n) * 32 + h * 8;
      vb.i[0] = *(const int4*)&vb_s[off];
      vb.i[1] = *(const int4*)&vb_s[off + 16];
      O[t].v = WMMA_F16(pa.v, vb.v, O[t].v);
    }
    __syncthreads();
  }

  // ---- epilogue: normalize into LDS (row pitch 65 -> conflict-free), then
  //      cooperative coalesced float4 residual + store ----
  #pragma unroll
  for (int r = 0; r < 8; ++r) {
    const float rinv = 1.0f / lrow[r];
    const int irel = (wv << 4) + r + 8 * h;
    #pragma unroll
    for (int t = 0; t < 4; ++t) {
      const int c = t * 16 + n;
      o_s[c * 65 + irel] = O[t].f[r] * rinv;
    }
  }
  __syncthreads();

  const float g = gamma[0];
  #pragma unroll
  for (int it = 0; it < 8; ++it) {
    const int idx = tid + it * 128;            // 1024 float4 slots
    const int c = idx >> 4, seg = idx & 15;
    const int i = seg << 2;
    const size_t off = (((size_t)(b * C + c)) << 12) + q0 + i;
    const float4 xr = *(const float4*)(x + off);
    float4 o;
    o.x = g * o_s[c * 65 + i + 0] + xr.x;
    o.y = g * o_s[c * 65 + i + 1] + xr.y;
    o.z = g * o_s[c * 65 + i + 2] + xr.z;
    o.w = g * o_s[c * 65 + i + 3] + xr.w;
    *(float4*)(out + off) = o;
  }
}

// ---------------------------------------------------------------------------

extern "C" void kernel_launch(void* const* d_in, const int* in_sizes, int n_in,
                              void* d_out, int out_size, void* d_ws, size_t ws_size,
                              hipStream_t stream) {
  (void)in_sizes; (void)n_in; (void)out_size; (void)ws_size;
  constexpr int B = 8, C = 64, L = 4096;

  const float* x     = (const float*)d_in[0];
  const float* Wq    = (const float*)d_in[1];
  const float* bq    = (const float*)d_in[2];
  const float* Wk    = (const float*)d_in[3];
  const float* bk    = (const float*)d_in[4];
  const float* Wv    = (const float*)d_in[5];
  const float* bv    = (const float*)d_in[6];
  const float* gamma = (const float*)d_in[7];
  float* out = (float*)d_out;

  _Float16* qT = (_Float16*)d_ws;                  // [B,L,C] f16, 4 MB
  _Float16* kT = qT + (size_t)B * L * C;           // [B,L,C] f16, 4 MB
  _Float16* vM = kT + (size_t)B * L * C;           // [B,C,L] f16, 4 MB

  qkv_proj_kernel<<<B * (L / 16), 128, 0, stream>>>(x, Wq, bq, Wk, bk, Wv, bv,
                                                    qT, kT, vM);
  attn_kernel<<<B * (L / 64), 128, 0, stream>>>(qT, kT, vM, x, gamma, out);
}
